// PatchEncoder_43310450213036
// MI455X (gfx1250) — compile-verified, matrix operand order checked by reading, not run on hardware
//
#include <hip/hip_runtime.h>
#include <hip/hip_bf16.h>
#include <stdint.h>

#define B_SZ   256
#define N_SEQ  196
#define P_DIM  768
#define D_DIM  768
#define N_MASK 40
#define N_UNM  156
#define KT     (P_DIM/32)        // 24 K-steps of 32
#define NTILES (D_DIM/16)        // 48 N-tiles

// byte strides in the packed-W layout: tile = (kt,nt), 1024 B per tile, lane gets 32 B
#define WPK_KT_STRIDE  (NTILES*1024)   // 49152 B per K-step
#define WPK_NT_STRIDE  1024

typedef __attribute__((ext_vector_type(16))) __bf16 v16bf;
typedef __attribute__((ext_vector_type(8)))  float  v8f;

union Frag16 { v16bf v; uint4 q[2]; };

// d_out element offsets (outputs concatenated flat, reference return order)
#define O_UE 0ll
#define O_ME 30670848ll            // 256*156*768
#define O_UP 38535168ll            // O_ME + 256*40*768
#define O_MI 69206016ll            // O_UP + 256*156*768
#define O_UI 69216256ll            // O_MI + 256*40

// workspace byte offsets
#define WS_WPK   0                 // packed bf16 W: 24*48*32*16 bf16 = 1,179,648 B
#define WS_MTOK  1179648           // 768 floats
#define WS_VALID (1179648+3072)    // 256 floats

// ---------------------------------------------------------------------------
// Repack W[P,D] fp32 row-major -> bf16 WMMA B-fragment order.
__global__ void pack_w_kernel(const float* __restrict__ W, __bf16* __restrict__ wpk) {
    int tile = blockIdx.x;                 // kt*NTILES + nt
    int kt = tile / NTILES, nt = tile % NTILES;
    int lane = threadIdx.x;
    int g = lane >> 4, i = lane & 15;
    int n = nt*16 + i;
    union { __bf16 e[16]; uint4 q[2]; } u;
#pragma unroll
    for (int e = 0; e < 16; ++e) {
        int k = kt*32 + 16*(e>>3) + 8*g + (e&7);
        u.e[e] = (__bf16)W[(size_t)k*D_DIM + n];
    }
    uint4* dst = (uint4*)(wpk + ((size_t)tile*32 + lane)*16);
    dst[0] = u.q[0];
    dst[1] = u.q[1];
}

// ---------------------------------------------------------------------------
// Per-image masking decisions. One block (8 waves) per image.
__global__ void mask_select_kernel(const float* __restrict__ patches,
                                   const float* __restrict__ rand_u,
                                   int*  __restrict__ mask_idx,
                                   int*  __restrict__ unmask_idx,
                                   float* __restrict__ valid_f) {
    int b = blockIdx.x;
    int t = threadIdx.x;
    int lane = t & 31, w = t >> 5;
    __shared__ float scores[N_SEQ];
    __shared__ int   validm[N_SEQ];
    __shared__ int   flag[N_SEQ];
    __shared__ int   s_valid_image;

    for (int n = w; n < N_SEQ; n += 8) {
        const float* row = patches + ((size_t)b*N_SEQ + n)*P_DIM;
        float acc = 0.f;
        for (int p = lane; p < P_DIM; p += 32) acc += fabsf(row[p]);
        for (int m = 16; m >= 1; m >>= 1) acc += __shfl_xor(acc, m, 32);
        if (lane == 0) {
            float mean = acc * (1.0f/(float)P_DIM);
            int v = (mean > 0.f) ? 1 : 0;
            float u = rand_u[b*N_SEQ + n];
            scores[n] = v ? (1000.f + 1000.f*u) : (10.f*u);
            validm[n] = v;
        }
    }
    __syncthreads();
    if (t == 0) {
        int vc = 0;
        for (int n = 0; n < N_SEQ; ++n) vc += validm[n];
        int max_masks = (int)ceilf((float)vc * 0.4f);
        s_valid_image = (vc >= 20 && max_masks >= N_MASK) ? 1 : 0;
    }
    __syncthreads();
    int valid_image = s_valid_image;

    if (t < N_SEQ) {
        float s = scores[t];
        int rank = 0;
        for (int m = 0; m < N_SEQ; ++m) {
            float sm = scores[m];
            rank += (sm > s) || (sm == s && m < t);
        }
        flag[t] = (rank < N_MASK) ? 1 : 0;
        if (valid_image && rank < N_MASK) mask_idx[b*N_MASK + rank] = t;
    }
    if (!valid_image && t < N_MASK) mask_idx[b*N_MASK + t] = 0;
    __syncthreads();

    if (valid_image) {
        if (t < N_SEQ && !flag[t]) {
            int u = 0;
            for (int m = 0; m < t; ++m) u += 1 - flag[m];
            if (u < N_UNM) unmask_idx[b*N_UNM + u] = t;
        }
    } else {
        if (t < N_UNM) unmask_idx[b*N_UNM + t] = t;
    }
    if (t == 0) valid_f[b] = valid_image ? 1.0f : 0.0f;
}

// ---------------------------------------------------------------------------
__global__ void mtok_kernel(const float* __restrict__ mask_token,
                            const float* __restrict__ W,
                            const float* __restrict__ bias,
                            float* __restrict__ mtok) {
    int d = blockIdx.x * blockDim.x + threadIdx.x;
    if (d >= D_DIM) return;
    float acc = bias[d];
    for (int p = 0; p < P_DIM; ++p) acc = fmaf(mask_token[p], W[(size_t)p*D_DIM + d], acc);
    mtok[d] = acc;
}

// ---------------------------------------------------------------------------
// Fused gather + GEMM + bias + pos epilogue, M-blocked.
// Block = (image b, group of NMT M-tiles): NMT*16 gathered rows in LDS (bf16).
// 8 waves; wave w owns N-tiles 6w..6w+5, processed in 2 passes of 3 so that
// accumulators stay at NMT*3 v8f. Each B fragment register is reused by NMT
// WMMAs (one per M-tile) -> B/L2 traffic scales as 1/NMT.
template<int NMT>
__global__ __launch_bounds__(256)
void gemm_gather_kernel(const float* __restrict__ patches,
                        const __bf16* __restrict__ wpk,
                        const float* __restrict__ bias,
                        const float* __restrict__ pos,
                        const int*   __restrict__ unmask_idx,
                        float* __restrict__ out_ue,
                        float* __restrict__ out_up,
                        int mt_base) {
    int b   = blockIdx.x;
    int mt0 = mt_base + blockIdx.y * NMT;
    int t = threadIdx.x;
    int lane = t & 31, w = t >> 5;

    // padded row stride: 776 bf16 = 388 dwords -> 4-bank skew (conflict-free)
    __shared__ __bf16 As[NMT*16][P_DIM + 8];
    __shared__ int    nsrc_sh[NMT*16];

    if (t < NMT*16) {
        int rowg = mt0*16 + t;
        nsrc_sh[t] = unmask_idx[b*N_UNM + (rowg < N_UNM ? rowg : N_UNM-1)];
    }
    __syncthreads();

    // stage NMT*16 gathered rows fp32 -> bf16 into LDS; wave w: rows w*2*NMT ..
#pragma unroll
    for (int rr = 0; rr < 2*NMT; ++rr) {
        int row  = w*2*NMT + rr;
        const float* src = patches + ((size_t)b*N_SEQ + nsrc_sh[row])*P_DIM + lane*4;
        __bf16* dst = &As[row][lane*4];
#pragma unroll
        for (int it = 0; it < P_DIM/128; ++it) {     // 6 iters of float4/lane
            float4 f = *(const float4*)(src + it*128);
            union { __bf16 e[4]; unsigned long long u64; } pk;
            pk.e[0] = (__bf16)f.x; pk.e[1] = (__bf16)f.y;
            pk.e[2] = (__bf16)f.z; pk.e[3] = (__bf16)f.w;
            *(unsigned long long*)(dst + it*128) = pk.u64;
        }
    }
    __syncthreads();

    int g = lane >> 4, i = lane & 15;
    const v8f vzero = {0.f,0.f,0.f,0.f,0.f,0.f,0.f,0.f};

    // per-M-tile LDS base pointers (ds offset immediates cover kt*64+32 <= 1504)
    const char* abasem[NMT];
#pragma unroll
    for (int m = 0; m < NMT; ++m)
        abasem[m] = (const char*)&As[m*16 + i][8*g];

#pragma unroll
    for (int pass = 0; pass < 2; ++pass) {
        const char* bbase = (const char*)wpk + (size_t)lane*32
                          + (size_t)(w*6 + pass*3)*WPK_NT_STRIDE;

        v8f acc[NMT][3];
#pragma unroll
        for (int m = 0; m < NMT; ++m)
#pragma unroll
            for (int j = 0; j < 3; ++j) acc[m][j] = vzero;

#pragma unroll 2
        for (int kt = 0; kt < KT; ++kt) {
            const char* bk = bbase + (size_t)kt*WPK_KT_STRIDE;

            // all 3 B fragments for this K-step in distinct registers
            Frag16 bm[3];
#pragma unroll
            for (int j = 0; j < 3; ++j) {
                bm[j].q[0] = *(const uint4*)(bk + j*WPK_NT_STRIDE);
                bm[j].q[1] = *(const uint4*)(bk + j*WPK_NT_STRIDE + 16);
            }

#pragma unroll
            for (int m = 0; m < NMT; ++m) {
                Frag16 a;
                a.q[0] = *(const uint4*)(abasem[m] + kt*64);
                a.q[1] = *(const uint4*)(abasem[m] + kt*64 + 32);
#pragma unroll
                for (int j = 0; j < 3; ++j) {
                    acc[m][j] = __builtin_amdgcn_wmma_f32_16x16x32_bf16(
                        false, a.v, false, bm[j].v, (short)0, acc[m][j], false, false);
                }
            }
        }

        // epilogue: C layout lane<16 -> N=lane, M=r ; lane>=16 -> N=lane-16, M=r+8
#pragma unroll
        for (int j = 0; j < 3; ++j) {
            int nt = w*6 + pass*3 + j;
            int d  = nt*16 + i;
            float bd = bias[d];
#pragma unroll
            for (int m = 0; m < NMT; ++m) {
#pragma unroll
                for (int r = 0; r < 8; ++r) {
                    int mm   = m*16 + 8*g + r;
                    int rowg = mt0*16 + mm;
                    if (rowg < N_UNM) {
                        int nsrc = nsrc_sh[mm];
                        float pv = pos[(size_t)nsrc*D_DIM + d];
                        size_t o = ((size_t)(b*N_UNM + rowg))*D_DIM + d;
                        out_ue[o] = acc[m][j][r] + bd + pv;
                        out_up[o] = pv;
                    }
                }
            }
        }
    }
}

// ---------------------------------------------------------------------------
// masked_embeddings[b,j,:] = (mtok + pos[mask_idx[b,j],:]) * validity[b]
__global__ void masked_emb_kernel(const float* __restrict__ mtok,
                                  const float* __restrict__ pos,
                                  const int*   __restrict__ mask_idx,
                                  const float* __restrict__ valid_f,
                                  float* __restrict__ out_me) {
    int b = blockIdx.x, j = blockIdx.y;
    int t = threadIdx.x;
    int mi = mask_idx[b*N_MASK + j];
    float vf = valid_f[b];
#pragma unroll
    for (int e = 0; e < 3; ++e) {
        int d = t + e*256;
        float v = (mtok[d] + pos[(size_t)mi*D_DIM + d]) * vf;
        out_me[((size_t)(b*N_MASK + j))*D_DIM + d] = v;
    }
}

// ---------------------------------------------------------------------------
extern "C" void kernel_launch(void* const* d_in, const int* in_sizes, int n_in,
                              void* d_out, int out_size, void* d_ws, size_t ws_size,
                              hipStream_t stream) {
    const float* patches = (const float*)d_in[0];
    const float* rand_u  = (const float*)d_in[1];
    const float* W       = (const float*)d_in[2];
    const float* bias    = (const float*)d_in[3];
    const float* pos     = (const float*)d_in[4];
    const float* mtoken  = (const float*)d_in[5];

    float* out      = (float*)d_out;
    float* out_ue   = out + O_UE;
    float* out_me   = out + O_ME;
    float* out_up   = out + O_UP;
    int*   mask_i   = (int*)(out + O_MI);
    int*   unmask_i = (int*)(out + O_UI);

    char*   ws      = (char*)d_ws;
    __bf16* wpk     = (__bf16*)(ws + WS_WPK);
    float*  mtok    = (float*)(ws + WS_MTOK);
    float*  valid_f = (float*)(ws + WS_VALID);

    // 1) repack W to bf16 WMMA fragment order
    pack_w_kernel<<<dim3(KT*NTILES), dim3(32), 0, stream>>>(W, wpk);

    // 2) per-image mask / unmask index selection (writes int32 outputs)
    mask_select_kernel<<<dim3(B_SZ), dim3(256), 0, stream>>>(
        patches, rand_u, mask_i, unmask_i, valid_f);

    // 3) mask-token embedding
    mtok_kernel<<<dim3(3), dim3(256), 0, stream>>>(mtoken, W, bias, mtok);

    // 4) fused gather + bf16 WMMA GEMM: M-tiles 0..7 (64 rows/block),
    //    then tail M-tiles 8..9 (32 rows/block)
    gemm_gather_kernel<4><<<dim3(B_SZ, 2), 256, 0, stream>>>(
        patches, wpk, bias, pos, unmask_i, out_ue, out_up, 0);
    gemm_gather_kernel<2><<<dim3(B_SZ, 1), 256, 0, stream>>>(
        patches, wpk, bias, pos, unmask_i, out_ue, out_up, 8);

    // 5) masked embeddings
    masked_emb_kernel<<<dim3(B_SZ, N_MASK), 256, 0, stream>>>(
        mtok, pos, mask_i, valid_f, out_me);
}
